// MultiHeadAttention_52286931861544
// MI455X (gfx1250) — matrix operand order repaired and checked by reading, not compiled, not run on hardware
//
#include <hip/hip_runtime.h>

// ---------------------------------------------------------------------------
// MI455X (gfx1250) fused multi-head attention, bf16 WMMA + flash attention.
// EMB=768, NH=12, DH=64, S=4096, B=1.
// ---------------------------------------------------------------------------

typedef __attribute__((ext_vector_type(16))) __bf16 v16bf;
typedef __attribute__((ext_vector_type(8)))  __bf16 v8bf;
typedef __attribute__((ext_vector_type(8)))  float  v8f;

union BF16x16 { v16bf v; v8bf h[2]; };

#define EMB_ 768
#define SEQ_ 4096
#define NH_  12
#define DH_  64

// ---------------------------------------------------------------------------
// DPP16 lane permutation (stays within each 16-lane row of the wave32).
// 0xB1 = quad_perm[1,0,3,2] (xor1), 0x4E = quad_perm[2,3,0,1] (xor2),
// 0x141 = row_half_mirror (pairs quads within 8), 0x140 = row_mirror
// (pairs 8-groups within 16).
// ---------------------------------------------------------------------------
template <int CTRL>
__device__ __forceinline__ float dpp_mov(float v) {
  int i = __builtin_bit_cast(int, v);
  int r = __builtin_amdgcn_update_dpp(0, i, CTRL, 0xf, 0xf, true);
  return __builtin_bit_cast(float, r);
}

__device__ __forceinline__ float rowmax16(float x) {
  x = fmaxf(x, dpp_mov<0xB1>(x));
  x = fmaxf(x, dpp_mov<0x4E>(x));
  x = fmaxf(x, dpp_mov<0x141>(x));
  x = fmaxf(x, dpp_mov<0x140>(x));
  return x;
}

__device__ __forceinline__ float rowsum16(float x) {
  x += dpp_mov<0xB1>(x);
  x += dpp_mov<0x4E>(x);
  x += dpp_mov<0x141>(x);
  x += dpp_mov<0x140>(x);
  return x;
}

// ---------------------------------------------------------------------------
// fp32 -> bf16 conversion
// ---------------------------------------------------------------------------
__global__ void cvt_f32_to_bf16(const float* __restrict__ in,
                                __bf16* __restrict__ out, int n) {
  int i = blockIdx.x * blockDim.x + threadIdx.x;
  if (i < n) out[i] = (__bf16)in[i];
}

// ---------------------------------------------------------------------------
// GEMM: C[M,N] = A[M,K] * W[N,K]^T + bias[N]
// block = 256 threads (8 waves of 32). Tile: 128 (M) x 64 (N), K step 32.
// W tile (64x32 bf16) staged in LDS, shared by all 8 waves.
// ---------------------------------------------------------------------------
template <bool F32OUT>
__global__ __launch_bounds__(256) void gemm_bf16(
    const __bf16* __restrict__ A, const __bf16* __restrict__ W,
    const float* __restrict__ bias, float* __restrict__ outF,
    __bf16* __restrict__ outB, int M, int N, int K) {
  __shared__ alignas(16) __bf16 Wl[64 * 32];

  const int t    = threadIdx.x;
  const int wid  = t >> 5;
  const int lane = t & 31;
  const int ln   = lane & 15;
  const int hf   = lane >> 4;
  const int mb   = blockIdx.x * 128;
  const int nb   = blockIdx.y * 64;
  const int mrow = mb + wid * 16 + ln;  // A-fragment row for this lane

  v8f acc[4] = {};

  for (int kk = 0; kk < K; kk += 32) {
    // Cooperative load: W tile [64 rows x 32 k] -> LDS, 16B per thread.
    {
      int r = t >> 2, c = t & 3;
      *(v8bf*)&Wl[r * 32 + c * 8] =
          *(const v8bf*)(W + (size_t)(nb + r) * K + kk + c * 8);
      if (kk + 32 < K) {  // prefetch next W tile (global_prefetch_b8)
        __builtin_prefetch(W + (size_t)(nb + r) * K + kk + 32 + c * 8, 0, 3);
      }
    }
    __syncthreads();

    // A fragment 16x32: lane row = ln, K chunks {hf*8..+7} and {16+hf*8..+7}
    BF16x16 af;
    const __bf16* ap = A + (size_t)mrow * K + kk + hf * 8;
    af.h[0] = *(const v8bf*)ap;
    af.h[1] = *(const v8bf*)(ap + 16);
    if (kk + 32 < K) __builtin_prefetch(ap + 32, 0, 3);

#pragma unroll
    for (int f = 0; f < 4; ++f) {
      // B fragment 32x16: lane col n = f*16+ln, K chunks {hf*8..+7} and {16+hf*8..+7}
      BF16x16 bf_;
      const __bf16* bp = &Wl[(f * 16 + ln) * 32 + hf * 8];
      bf_.h[0] = *(const v8bf*)bp;
      bf_.h[1] = *(const v8bf*)(bp + 16);
      acc[f] = __builtin_amdgcn_wmma_f32_16x16x32_bf16(
          false, af.v, false, bf_.v, (short)0, acc[f], false, false);
    }
    __syncthreads();
  }

#pragma unroll
  for (int f = 0; f < 4; ++f) {
    int col = nb + f * 16 + ln;
    float b = bias[col];
#pragma unroll
    for (int g = 0; g < 8; ++g) {
      int row = mb + wid * 16 + g + 8 * hf;  // C layout: row = g + 8*half
      float v = acc[f][g] + b;
      if (F32OUT)
        outF[(size_t)row * N + col] = v;
      else
        outB[(size_t)row * N + col] = (__bf16)v;
    }
  }
}

// ---------------------------------------------------------------------------
// Flash attention: one 16-query tile per wave, 4 waves/block, 32 keys/iter.
// grid = (S/64, NH). K tile row-major + V tile transposed staged in LDS.
// ---------------------------------------------------------------------------
__global__ __launch_bounds__(128) void flash_attn(
    const __bf16* __restrict__ Q, const __bf16* __restrict__ Kg,
    const __bf16* __restrict__ V, const float* __restrict__ mask,
    __bf16* __restrict__ O) {
  __shared__ alignas(16) __bf16 Kl[32 * 64];        // [key][dh]
  __shared__ alignas(16) __bf16 Vt[64 * 32];        // [dh][key] (transposed)
  __shared__ alignas(16) __bf16 Pl[4][16 * 32];     // per-wave P tile

  const int t    = threadIdx.x;
  const int wid  = t >> 5;
  const int lane = t & 31;
  const int ln   = lane & 15;
  const int hf   = lane >> 4;
  const int h    = blockIdx.y;
  const int qb   = blockIdx.x * 64 + wid * 16;
  const int hc   = h * DH_;

  // Q fragments: two 16x32 A-fragments covering dh 0..31 / 32..63 (persist).
  BF16x16 qf[2];
#pragma unroll
  for (int qh = 0; qh < 2; ++qh) {
    const __bf16* p = Q + (size_t)(qb + ln) * EMB_ + hc + qh * 32 + hf * 8;
    qf[qh].h[0] = *(const v8bf*)p;
    qf[qh].h[1] = *(const v8bf*)(p + 16);
  }

  v8f o[4] = {};
  float m[8], l[8];
#pragma unroll
  for (int g = 0; g < 8; ++g) { m[g] = -3.0e38f; l[g] = 0.0f; }

  for (int kb = 0; kb < SEQ_; kb += 32) {
    // Cooperative stage: K row-major, V transposed. 256 16B-chunks total.
#pragma unroll
    for (int it = 0; it < 2; ++it) {
      int idx = t + it * 128;
      int r = idx >> 3, c = idx & 7;  // key row r (0..31), 8-elem chunk c
      *(v8bf*)&Kl[r * 64 + c * 8] =
          *(const v8bf*)(Kg + (size_t)(kb + r) * EMB_ + hc + c * 8);
      v8bf vv = *(const v8bf*)(V + (size_t)(kb + r) * EMB_ + hc + c * 8);
#pragma unroll
      for (int j = 0; j < 8; ++j) Vt[(c * 8 + j) * 32 + r] = vv[j];
    }
    // Prefetch next iteration's K/V rows (one cacheline per lane).
    if (kb + 32 < SEQ_) {
      int r = t & 31;
      __builtin_prefetch(Kg + (size_t)(kb + 32 + r) * EMB_ + hc, 0, 3);
      __builtin_prefetch(V  + (size_t)(kb + 32 + r) * EMB_ + hc, 0, 3);
    }
    __syncthreads();

    // scores = Q (16x64) @ K^T (64x32): two 16x16 accumulators, K-dim chained
    v8f sc[2];
#pragma unroll
    for (int ns = 0; ns < 2; ++ns) {
      v8f z = {};
#pragma unroll
      for (int kh = 0; kh < 2; ++kh) {
        BF16x16 bf_;  // B[dh][key]: lane key = ns*16+ln, dh chunks {hf*8..+7},{16+hf*8..+7}
        const __bf16* p = &Kl[(ns * 16 + ln) * 64 + kh * 32 + hf * 8];
        bf_.h[0] = *(const v8bf*)p;
        bf_.h[1] = *(const v8bf*)(p + 16);
        z = __builtin_amdgcn_wmma_f32_16x16x32_bf16(
            false, qf[kh].v, false, bf_.v, (short)0, z, false, false);
      }
      sc[ns] = z;
    }

    const float mt0 = -10000.0f * (1.0f - mask[kb + ln]);
    const float mt1 = -10000.0f * (1.0f - mask[kb + 16 + ln]);

    // Online softmax. C layout: reg g, lane -> row g+8*hf, col ln.
    // Row reductions via DPP16 butterflies (stay within 16-lane rows).
#pragma unroll
    for (int g = 0; g < 8; ++g) {
      float s0 = sc[0][g] * 0.125f + mt0;
      float s1 = sc[1][g] * 0.125f + mt1;
      float mx = rowmax16(fmaxf(s0, s1));
      float nm = fmaxf(m[g], mx);
      float al = __expf(m[g] - nm);
      m[g] = nm;
      float p0 = __expf(s0 - nm);
      float p1 = __expf(s1 - nm);
      float rs = rowsum16(p0 + p1);
      l[g] = l[g] * al + rs;
      // re-layout P: C-fragment -> row-major bf16 in per-wave LDS
      int row = g + 8 * hf;
      Pl[wid][row * 32 + ln]      = (__bf16)p0;
      Pl[wid][row * 32 + 16 + ln] = (__bf16)p1;
#pragma unroll
      for (int f = 0; f < 4; ++f) o[f][g] *= al;
    }

    // P fragment (A 16x32): lane row = ln, chunked K through LDS.
    BF16x16 pf;
    {
      const __bf16* p = &Pl[wid][ln * 32 + hf * 8];
      pf.h[0] = *(const v8bf*)p;
      pf.h[1] = *(const v8bf*)(p + 16);
    }
    // O += P (16x32) @ V (32x64): 4 WMMAs over dh sub-tiles.
#pragma unroll
    for (int f = 0; f < 4; ++f) {
      BF16x16 vf;  // B[key][dh]: lane dh = f*16+ln, key chunks {hf*8..+7},{16+hf*8..+7}
      const __bf16* p = &Vt[(f * 16 + ln) * 32 + hf * 8];
      vf.h[0] = *(const v8bf*)p;
      vf.h[1] = *(const v8bf*)(p + 16);
      o[f] = __builtin_amdgcn_wmma_f32_16x16x32_bf16(
          false, pf.v, false, vf.v, (short)0, o[f], false, false);
    }
    __syncthreads();
  }

  // Finalize: O /= l, store bf16 head output.
#pragma unroll
  for (int g = 0; g < 8; ++g) {
    float inv = 1.0f / l[g];
    int row = qb + g + 8 * hf;
#pragma unroll
    for (int f = 0; f < 4; ++f)
      O[(size_t)row * EMB_ + hc + f * 16 + ln] = (__bf16)(o[f][g] * inv);
  }
}

// ---------------------------------------------------------------------------
// Launch
// ---------------------------------------------------------------------------
extern "C" void kernel_launch(void* const* d_in, const int* in_sizes, int n_in,
                              void* d_out, int out_size, void* d_ws,
                              size_t ws_size, hipStream_t stream) {
  const float* x    = (const float*)d_in[0];
  const float* mask = (const float*)d_in[1];
  const float* Wq   = (const float*)d_in[2];
  const float* bq   = (const float*)d_in[3];
  const float* Wk   = (const float*)d_in[4];
  const float* bk   = (const float*)d_in[5];
  const float* Wv   = (const float*)d_in[6];
  const float* bv   = (const float*)d_in[7];
  const float* Wo   = (const float*)d_in[8];
  const float* bo   = (const float*)d_in[9];
  float* out = (float*)d_out;

  const size_t XE = (size_t)SEQ_ * EMB_;   // 3,145,728
  const size_t WE = (size_t)EMB_ * EMB_;   // 589,824

  char* ws = (char*)d_ws;
  size_t off = 0;
  auto alloc = [&](size_t bytes) {
    void* p = ws + off;
    off += (bytes + 255) & ~(size_t)255;
    return p;
  };
  __bf16* xbf = (__bf16*)alloc(XE * 2);
  __bf16* Wqb = (__bf16*)alloc(WE * 2);
  __bf16* Wkb = (__bf16*)alloc(WE * 2);
  __bf16* Wvb = (__bf16*)alloc(WE * 2);
  __bf16* Wob = (__bf16*)alloc(WE * 2);
  __bf16* Qb  = (__bf16*)alloc(XE * 2);
  __bf16* Kb  = (__bf16*)alloc(XE * 2);
  __bf16* Vb  = (__bf16*)alloc(XE * 2);
  __bf16* Ab  = (__bf16*)alloc(XE * 2);

  // Stage 1: fp32 -> bf16
  cvt_f32_to_bf16<<<(int)((XE + 255) / 256), 256, 0, stream>>>(x, xbf, (int)XE);
  cvt_f32_to_bf16<<<(int)((WE + 255) / 256), 256, 0, stream>>>(Wq, Wqb, (int)WE);
  cvt_f32_to_bf16<<<(int)((WE + 255) / 256), 256, 0, stream>>>(Wk, Wkb, (int)WE);
  cvt_f32_to_bf16<<<(int)((WE + 255) / 256), 256, 0, stream>>>(Wv, Wvb, (int)WE);
  cvt_f32_to_bf16<<<(int)((WE + 255) / 256), 256, 0, stream>>>(Wo, Wob, (int)WE);

  // Stage 2: QKV projections (bf16 out)
  dim3 gg(SEQ_ / 128, EMB_ / 64);
  gemm_bf16<false><<<gg, 256, 0, stream>>>(xbf, Wqb, bq, nullptr, Qb,
                                           SEQ_, EMB_, EMB_);
  gemm_bf16<false><<<gg, 256, 0, stream>>>(xbf, Wkb, bk, nullptr, Kb,
                                           SEQ_, EMB_, EMB_);
  gemm_bf16<false><<<gg, 256, 0, stream>>>(xbf, Wvb, bv, nullptr, Vb,
                                           SEQ_, EMB_, EMB_);

  // Stage 3: flash attention (never materializes S x S scores)
  flash_attn<<<dim3(SEQ_ / 64, NH_), 128, 0, stream>>>(Qb, Kb, Vb, mask, Ab);

  // Stage 4: output projection (fp32 out)
  gemm_bf16<true><<<gg, 256, 0, stream>>>(Ab, Wob, bo, out, nullptr,
                                          SEQ_, EMB_, EMB_);
}
